// TFTransfoXLMainLayer_41377714930141
// MI455X (gfx1250) — matrix operand order repaired and checked by reading, not compiled, not run on hardware
//
#include <hip/hip_runtime.h>
#include <hip/hip_bf16.h>

typedef __attribute__((ext_vector_type(16))) _Float16 v16h;
typedef __attribute__((ext_vector_type(8)))  float    v8f;

// ---------------- model dims ----------------
#define QL   512
#define ML   512
#define KL   1024
#define BS   4
#define DM   1024
#define NH   16
#define DH   64
#define DI   4096
#define NL   4

union FragU { uint4 u[2]; v16h v; };
union PackU { _Float16 s[16]; uint4 u[2]; };

// ============================================================
// Generic batched f16 WMMA GEMM:  C[batch] = A[batch] * B[batch % bmod]
// A: MxK row-major (lda), B: KxN row-major (ldb), C: MxN f32 (ldc)
// Block tile 128x128, K-step 32. 8 waves, each wave computes 32x64
// (2x4 grid of 16x16 v_wmma_f32_16x16x32_f16 accumulators).
// Out-of-range staging rows/cols are CLAMPED (not masked): a clamped
// A-row/B-col only feeds C entries that the guarded store discards,
// so the hot loop carries no exec-mask manipulation at all.
// ============================================================
__global__ __launch_bounds__(256) void k_gemm_f16(
    const _Float16* __restrict__ A, const _Float16* __restrict__ B,
    float* __restrict__ C, int M, int N, int K,
    int lda, int ldb, int ldc,
    long long sA, long long sB, long long sC, int bmod)
{
    __shared__ _Float16 sa[128 * 32];   // [m][k]
    __shared__ _Float16 sb[128 * 32];   // [n][k]  (transposed on fill)
    const int bz = blockIdx.z;
    A += (long long)bz * sA;
    B += (long long)(bz % bmod) * sB;
    C += (long long)bz * sC;

    const int m0   = blockIdx.y * 128;
    const int n0   = blockIdx.x * 128;
    const int tid  = threadIdx.x;
    const int lane = tid & 31;
    const int w    = tid >> 5;
    const int wm   = (w >> 1) << 5;   // 4 waves along M
    const int wn   = (w & 1) << 6;    // 2 waves along N

    v8f acc[2][4];
#pragma unroll
    for (int i = 0; i < 2; ++i)
#pragma unroll
        for (int j = 0; j < 4; ++j)
#pragma unroll
            for (int e = 0; e < 8; ++e) acc[i][j][e] = 0.0f;

    const int mrow  = tid >> 1;          // 0..127  (A stage row)
    const int koffA = (tid & 1) << 4;    // 0 or 16 halves
    const int nlB   = tid & 127;         // B stage: this thread's n-column
    const int khB   = (tid >> 7) << 4;   // 0 or 16: K half this thread covers
    const int hi    = lane >> 4;         // half-wave select
    const int lml   = lane & 15;

    // clamped staging addresses (see note above)
    const int mC = min(m0 + mrow, M - 1);
    const int nC = min(n0 + nlB, N - 1);
    const _Float16* aPtr = A + (size_t)mC * lda + koffA;
    const _Float16* bPtr = B + (size_t)khB * (size_t)ldb + nC;
    const size_t bStep = (size_t)32 * (size_t)ldb;

    for (int k0 = 0; k0 < K; k0 += 32) {
        // ---- stage A tile (coalesced 32B per thread) ----
        uint4 a0 = ((const uint4*)aPtr)[0];
        uint4 a1 = ((const uint4*)aPtr)[1];

        // ---- stage B tile transposed to [n][k]: one n-column per thread ----
        PackU bt;
        {
            const _Float16* bp = bPtr;
#pragma unroll
            for (int i = 0; i < 16; ++i) {
                bt.s[i] = *bp;
                bp += ldb;
            }
        }
        aPtr += 32;
        bPtr += bStep;
        __builtin_prefetch(aPtr, 0, 0);      // global_prefetch_b8 next tiles
        __builtin_prefetch(bPtr, 0, 0);

        *(uint4*)(&sa[mrow * 32 + koffA])     = a0;
        *(uint4*)(&sa[mrow * 32 + koffA + 8]) = a1;
        *(uint4*)(&sb[nlB * 32 + khB])        = bt.u[0];
        *(uint4*)(&sb[nlB * 32 + khB + 8])    = bt.u[1];
        __syncthreads();

        // ---- load fragments from LDS ----
        FragU af[2], bf[4];
#pragma unroll
        for (int mt = 0; mt < 2; ++mt) {
            const _Float16* base = &sa[(wm + mt * 16 + lml) * 32 + hi * 8];
            af[mt].u[0] = *(const uint4*)(base);
            af[mt].u[1] = *(const uint4*)(base + 16);
        }
#pragma unroll
        for (int nt = 0; nt < 4; ++nt) {
            const _Float16* base = &sb[(wn + nt * 16 + lml) * 32 + hi * 8];
            bf[nt].u[0] = *(const uint4*)(base);
            bf[nt].u[1] = *(const uint4*)(base + 16);
        }

#pragma unroll
        for (int mt = 0; mt < 2; ++mt)
#pragma unroll
            for (int nt = 0; nt < 4; ++nt)
                acc[mt][nt] = __builtin_amdgcn_wmma_f32_16x16x32_f16(
                    false, af[mt].v, false, bf[nt].v, (short)0, acc[mt][nt], false, false);
        __syncthreads();
    }

    // ---- store C (f32 16x16 layout: elem r -> m = r + 8*hi, n = lane&15) ----
#pragma unroll
    for (int mt = 0; mt < 2; ++mt)
#pragma unroll
        for (int nt = 0; nt < 4; ++nt)
#pragma unroll
            for (int r = 0; r < 8; ++r) {
                int m = m0 + wm + mt * 16 + (hi << 3) + r;
                int n = n0 + wn + nt * 16 + lml;
                if (m < M && n < N) C[(size_t)m * ldc + n] = acc[mt][nt][r];
            }
}

// ============================================================
// Elementwise / pack / reduction kernels
// ============================================================
__global__ __launch_bounds__(256) void k_f32_to_h(const float* __restrict__ s,
                                                  _Float16* __restrict__ d, int n)
{
    int i = blockIdx.x * 256 + threadIdx.x;
    if (i < n) d[i] = (_Float16)s[i];
}

// adaptive embedding: one block per flat token (f = i*BS + b)
__global__ __launch_bounds__(256) void k_embed(
    const int* __restrict__ ids,
    const float* __restrict__ e0, const float* __restrict__ p0,
    const float* __restrict__ e1, const float* __restrict__ p1,
    const float* __restrict__ e2, const float* __restrict__ p2,
    const float* __restrict__ e3, const float* __restrict__ p3,
    float* __restrict__ h)
{
    __shared__ float sh[1024];
    const int f  = blockIdx.x;         // 0..2047
    const int b  = f & 3, i = f >> 2;
    const int id = ids[b * QL + i];
    const float* tab; const float* proj; int d, off;
    if (id < 20000)       { tab = e0; proj = p0; d = 1024; off = 0; }
    else if (id < 40000)  { tab = e1; proj = p1; d = 256;  off = 20000; }
    else if (id < 200000) { tab = e2; proj = p2; d = 64;   off = 40000; }
    else                  { tab = e3; proj = p3; d = 16;   off = 200000; }
    const int t = threadIdx.x;
    for (int dd = t; dd < d; dd += 256) sh[dd] = tab[(size_t)(id - off) * d + dd];
    __syncthreads();
#pragma unroll
    for (int q = 0; q < 4; ++q) {
        int c = t + q * 256;
        float s = 0.0f;
        for (int dd = 0; dd < d; ++dd) s += sh[dd] * proj[(size_t)dd * DM + c];
        h[(size_t)f * DM + c] = s * 32.0f;   // sqrt(DM)
    }
}

__global__ __launch_bounds__(256) void k_posemb(float* __restrict__ pe,
                                                _Float16* __restrict__ peh)
{
    int idx = blockIdx.x * 256 + threadIdx.x;
    if (idx >= KL * DM) return;
    int p = idx >> 10, c = idx & 1023;
    int cc = c & 511;
    float inv = __expf(-logf(10000.0f) * (float)(2 * cc) / 1024.0f);
    float val = (float)(KL - 1 - p) * inv;
    float r = (c < 512) ? __sinf(val) : __cosf(val);
    pe[idx] = r;
    peh[idx] = (_Float16)r;
}

// cat = concat(mems[l], h) -> (klen*BS, DM) f16, row = j*BS+b
__global__ __launch_bounds__(256) void k_cat(const float* __restrict__ mems_l,
                                             const float* __restrict__ h,
                                             _Float16* __restrict__ cat)
{
    int idx = blockIdx.x * 256 + threadIdx.x;  // over 4096*1024
    int c = idx & 1023, row = idx >> 10;
    int b = row & 3, j = row >> 2;
    float v = (j < ML) ? mems_l[((size_t)j * BS + b) * DM + c]
                       : h[((size_t)(j - ML) * BS + b) * DM + c];
    cat[idx] = (_Float16)v;
}

// q packs with both biases: layout (b,n,i,d)
__global__ __launch_bounds__(256) void k_pack_q(const float* __restrict__ wh,
                                                const float* __restrict__ rwb,
                                                const float* __restrict__ rrb,
                                                _Float16* __restrict__ qrw,
                                                _Float16* __restrict__ qrr)
{
    int idx = blockIdx.x * 256 + threadIdx.x;   // 64*512*64
    int bn = idx >> 15, rem = idx & 32767;
    int i = rem >> 6, d = rem & 63;
    int b = bn >> 4, n = bn & 15;
    float q = wh[((size_t)(i + ML) * BS + b) * (3 * DM) + n * DH + d];
    qrw[idx] = (_Float16)(q + rwb[n * DH + d]);
    qrr[idx] = (_Float16)(q + rrb[n * DH + d]);
}

// kT: (b,n,d,j)
__global__ __launch_bounds__(256) void k_pack_k(const float* __restrict__ wh,
                                                _Float16* __restrict__ kT)
{
    int idx = blockIdx.x * 256 + threadIdx.x;   // 64*64*1024
    int bn = idx >> 16, rem = idx & 65535;
    int d = rem >> 10, j = rem & 1023;
    int b = bn >> 4, n = bn & 15;
    kT[idx] = (_Float16)wh[((size_t)j * BS + b) * (3 * DM) + DM + n * DH + d];
}

// v: (b,n,j,d)
__global__ __launch_bounds__(256) void k_pack_v(const float* __restrict__ wh,
                                                _Float16* __restrict__ vh)
{
    int idx = blockIdx.x * 256 + threadIdx.x;   // 64*1024*64
    int bn = idx >> 16, rem = idx & 65535;
    int j = rem >> 6, d = rem & 63;
    int b = bn >> 4, n = bn & 15;
    vh[idx] = (_Float16)wh[((size_t)j * BS + b) * (3 * DM) + 2 * DM + n * DH + d];
}

// rkT: (n,d,j)
__global__ __launch_bounds__(256) void k_pack_rk(const float* __restrict__ rk,
                                                 _Float16* __restrict__ rkT)
{
    int idx = blockIdx.x * 256 + threadIdx.x;   // 16*64*1024
    int n = idx >> 16, rem = idx & 65535;
    int d = rem >> 10, j = rem & 1023;
    rkT[idx] = (_Float16)rk[(size_t)j * DM + n * DH + d];
}

// softmax over klen with fused rel_shift and causal mask. one (n,i) row/block.
__global__ __launch_bounds__(256) void k_softmax(const float* __restrict__ AC,
                                                 const float* __restrict__ BD,
                                                 _Float16* __restrict__ prob)
{
    const int i = blockIdx.x, n = blockIdx.y, t = threadIdx.x;
    const float scale = 0.125f;   // 1/sqrt(64)
    const size_t base = ((size_t)n * QL + i) * (size_t)KL;
    float v[4];
    float mx = -3.0e38f;
#pragma unroll
    for (int q = 0; q < 4; ++q) {
        int j = t + q * 256;
        float s;
        if (j <= i + ML) {
            int jj = j + (QL - 1) - i;                   // rel_shift
            float bd = (jj < KL) ? BD[base + jj] : 0.0f;
            s = (AC[base + j] + bd) * scale;
        } else {
            s = -3.0e38f;
        }
        v[q] = s;
        mx = fmaxf(mx, s);
    }
    __shared__ float red[256];
    red[t] = mx; __syncthreads();
    for (int o = 128; o > 0; o >>= 1) { if (t < o) red[t] = fmaxf(red[t], red[t + o]); __syncthreads(); }
    mx = red[0]; __syncthreads();
    float sum = 0.0f;
#pragma unroll
    for (int q = 0; q < 4; ++q) {
        v[q] = (v[q] > -1.0e37f) ? __expf(v[q] - mx) : 0.0f;
        sum += v[q];
    }
    red[t] = sum; __syncthreads();
    for (int o = 128; o > 0; o >>= 1) { if (t < o) red[t] += red[t + o]; __syncthreads(); }
    float inv = 1.0f / red[0];
#pragma unroll
    for (int q = 0; q < 4; ++q)
        prob[base + t + q * 256] = (_Float16)(v[q] * inv);
}

// attn (n,i,d) for batch b -> attnrow[(i*BS+b)][n*DH+d] f16
__global__ __launch_bounds__(256) void k_pack_attn(const float* __restrict__ attn,
                                                   _Float16* __restrict__ attnrow, int b)
{
    int idx = blockIdx.x * 256 + threadIdx.x;   // 16*512*64
    int n = idx >> 15, rem = idx & 32767;
    int i = rem >> 6, d = rem & 63;
    attnrow[((size_t)i * BS + b) * DM + n * DH + d] = (_Float16)attn[idx];
}

// h = LayerNorm(h + add (+ bias)) ; writes f32 and f16 copies. one row/block.
__global__ __launch_bounds__(256) void k_add_ln(float* __restrict__ h,
                                                const float* __restrict__ add,
                                                const float* __restrict__ bias,
                                                const float* __restrict__ g,
                                                const float* __restrict__ bb,
                                                _Float16* __restrict__ hh)
{
    const int row = blockIdx.x, t = threadIdx.x;
    float x[4];
    float s = 0.0f, s2 = 0.0f;
#pragma unroll
    for (int q = 0; q < 4; ++q) {
        int c = t + q * 256;
        float v = h[(size_t)row * DM + c] + add[(size_t)row * DM + c];
        if (bias) v += bias[c];
        x[q] = v; s += v; s2 += v * v;
    }
    __shared__ float r1[256], r2[256];
    r1[t] = s; r2[t] = s2; __syncthreads();
    for (int o = 128; o > 0; o >>= 1) {
        if (t < o) { r1[t] += r1[t + o]; r2[t] += r2[t + o]; }
        __syncthreads();
    }
    float mean = r1[0] * (1.0f / DM);
    float var  = r2[0] * (1.0f / DM) - mean * mean;
    float rinv = rsqrtf(var + 1e-5f);
#pragma unroll
    for (int q = 0; q < 4; ++q) {
        int c = t + q * 256;
        float y = (x[q] - mean) * rinv * g[c] + bb[c];
        h[(size_t)row * DM + c]  = y;
        hh[(size_t)row * DM + c] = (_Float16)y;
    }
}

// cols must be a power of two (DI = 4096): colmask = DI-1
__global__ __launch_bounds__(256) void k_bias_relu_h(const float* __restrict__ s,
                                                     const float* __restrict__ bias,
                                                     _Float16* __restrict__ d,
                                                     int colmask, int n)
{
    int i = blockIdx.x * 256 + threadIdx.x;
    if (i >= n) return;
    float v = s[i] + bias[i & colmask];
    d[i] = (_Float16)fmaxf(v, 0.0f);
}

// (i*BS+b, c) -> (b, i, c)
__global__ __launch_bounds__(256) void k_out(const float* __restrict__ h,
                                             float* __restrict__ out)
{
    int idx = blockIdx.x * 256 + threadIdx.x;   // 2048*1024
    int c = idx & 1023, row = idx >> 10;
    int b = row & 3, i = row >> 2;
    out[((size_t)b * QL + i) * DM + c] = h[idx];
}

// ============================================================
// Host orchestration
// ============================================================
extern "C" void kernel_launch(void* const* d_in, const int* in_sizes, int n_in,
                              void* d_out, int out_size, void* d_ws, size_t ws_size,
                              hipStream_t stream)
{
    (void)in_sizes; (void)n_in; (void)out_size; (void)ws_size;
    const int*   ids   = (const int*)  d_in[0];
    const float* mems  = (const float*)d_in[1];
    const float* e0    = (const float*)d_in[2];
    const float* p0    = (const float*)d_in[3];
    const float* e1    = (const float*)d_in[4];
    const float* p1    = (const float*)d_in[5];
    const float* e2    = (const float*)d_in[6];
    const float* p2    = (const float*)d_in[7];
    const float* e3    = (const float*)d_in[8];
    const float* p3    = (const float*)d_in[9];
    const float* qkvw  = (const float*)d_in[10];
    const float* rnetw = (const float*)d_in[11];
    const float* ow    = (const float*)d_in[12];
    const float* rwb   = (const float*)d_in[13];
    const float* rrb   = (const float*)d_in[14];
    const float* ln1g  = (const float*)d_in[15];
    const float* ln1b  = (const float*)d_in[16];
    const float* ffw1  = (const float*)d_in[17];
    const float* ffb1  = (const float*)d_in[18];
    const float* ffw2  = (const float*)d_in[19];
    const float* ffb2  = (const float*)d_in[20];
    const float* ln2g  = (const float*)d_in[21];
    const float* ln2b  = (const float*)d_in[22];

    // ---- workspace carve ----
    char* wp = (char*)d_ws;
    auto alloc = [&](size_t bytes) -> void* {
        void* r = (void*)wp;
        wp += (bytes + 255) & ~(size_t)255;
        return r;
    };
    _Float16* Wqkv  = (_Float16*)alloc((size_t)NL * DM * 3 * DM * 2);
    _Float16* Wrnet = (_Float16*)alloc((size_t)NL * DM * DM * 2);
    _Float16* Wo    = (_Float16*)alloc((size_t)NL * DM * DM * 2);
    _Float16* Wff1  = (_Float16*)alloc((size_t)NL * DM * DI * 2);
    _Float16* Wff2  = (_Float16*)alloc((size_t)NL * DI * DM * 2);
    float*    h     = (float*)   alloc((size_t)QL * BS * DM * 4);
    _Float16* hh    = (_Float16*)alloc((size_t)QL * BS * DM * 2);
    _Float16* cat   = (_Float16*)alloc((size_t)KL * BS * DM * 2);
    float*    wh    = (float*)   alloc((size_t)KL * BS * 3 * DM * 4);
    float*    pe    = (float*)   alloc((size_t)KL * DM * 4);
    _Float16* peh   = (_Float16*)alloc((size_t)KL * DM * 2);
    float*    rkbuf = (float*)   alloc((size_t)KL * DM * 4);
    _Float16* qrw   = (_Float16*)alloc((size_t)BS * NH * QL * DH * 2);
    _Float16* qrr   = (_Float16*)alloc((size_t)BS * NH * QL * DH * 2);
    _Float16* kT    = (_Float16*)alloc((size_t)BS * NH * DH * KL * 2);
    _Float16* vh    = (_Float16*)alloc((size_t)BS * NH * KL * DH * 2);
    _Float16* rkT   = (_Float16*)alloc((size_t)NH * DH * KL * 2);
    float*    AC    = (float*)   alloc((size_t)NH * QL * KL * 4);
    float*    BD    = (float*)   alloc((size_t)NH * QL * KL * 4);
    _Float16* prob  = (_Float16*)alloc((size_t)NH * QL * KL * 2);
    float*    attn  = (float*)   alloc((size_t)NH * QL * DH * 4);
    _Float16* arow  = (_Float16*)alloc((size_t)QL * BS * DM * 2);
    float*    gout  = (float*)   alloc((size_t)QL * BS * DM * 4);
    float*    core1 = (float*)   alloc((size_t)QL * BS * DI * 4);
    _Float16* core1h= (_Float16*)alloc((size_t)QL * BS * DI * 2);

    auto conv = [&](const float* s, _Float16* d, int n) {
        k_f32_to_h<<<(n + 255) / 256, 256, 0, stream>>>(s, d, n);
    };
    auto gemm = [&](const _Float16* A, const _Float16* B, float* C,
                    int M, int N, int K, int lda, int ldb, int ldc,
                    long long sA, long long sB, long long sC, int bmod, int batches) {
        dim3 g((N + 127) / 128, (M + 127) / 128, batches);
        k_gemm_f16<<<g, 256, 0, stream>>>(A, B, C, M, N, K, lda, ldb, ldc, sA, sB, sC, bmod);
    };

    // ---- weight conversion (f32 -> f16, all layers) ----
    conv(qkvw,  Wqkv,  NL * DM * 3 * DM);
    conv(rnetw, Wrnet, NL * DM * DM);
    conv(ow,    Wo,    NL * DM * DM);
    conv(ffw1,  Wff1,  NL * DM * DI);
    conv(ffw2,  Wff2,  NL * DI * DM);

    k_posemb<<<(KL * DM) / 256, 256, 0, stream>>>(pe, peh);
    k_embed<<<QL * BS, 256, 0, stream>>>(ids, e0, p0, e1, p1, e2, p2, e3, p3, h);

    for (int l = 0; l < NL; ++l) {
        k_cat<<<(KL * BS * DM) / 256, 256, 0, stream>>>(mems + (size_t)l * ML * BS * DM, h, cat);

        // w_heads = cat @ qkv_w[l] : (4096 x 1024) @ (1024 x 3072)
        gemm(cat, Wqkv + (size_t)l * DM * 3 * DM, wh,
             KL * BS, 3 * DM, DM, DM, 3 * DM, 3 * DM, 0, 0, 0, 1, 1);
        // rk = pos_emb @ r_net_w[l] : (1024 x 1024) @ (1024 x 1024)
        gemm(peh, Wrnet + (size_t)l * DM * DM, rkbuf,
             KL, DM, DM, DM, DM, DM, 0, 0, 0, 1, 1);

        k_pack_q<<<(BS * NH * QL * DH) / 256, 256, 0, stream>>>(
            wh, rwb + (size_t)l * NH * DH, rrb + (size_t)l * NH * DH, qrw, qrr);
        k_pack_k<<<(BS * NH * DH * KL) / 256, 256, 0, stream>>>(wh, kT);
        k_pack_v<<<(BS * NH * KL * DH) / 256, 256, 0, stream>>>(wh, vh);
        k_pack_rk<<<(NH * DH * KL) / 256, 256, 0, stream>>>(rkbuf, rkT);

        for (int b = 0; b < BS; ++b) {
            const long long qs = (long long)QL * DH;     // per-head A stride
            const long long ks = (long long)DH * KL;     // per-head kT/rkT stride
            // AC[n] = (q + r_w_bias) @ k^T  : 16 batches of (512x64)@(64x1024)
            gemm(qrw + (size_t)b * NH * QL * DH, kT + (size_t)b * NH * DH * KL, AC,
                 QL, KL, DH, DH, KL, KL, qs, ks, (long long)QL * KL, 16, NH);
            // BDraw[n] = (q + r_r_bias) @ rk^T (rk shared over b)
            gemm(qrr + (size_t)b * NH * QL * DH, rkT, BD,
                 QL, KL, DH, DH, KL, KL, qs, ks, (long long)QL * KL, 16, NH);

            k_softmax<<<dim3(QL, NH), 256, 0, stream>>>(AC, BD, prob);

            // attn[n] = prob @ v : 16 batches of (512x1024)@(1024x64)
            gemm(prob, vh + (size_t)b * NH * KL * DH, attn,
                 QL, DH, KL, KL, DH, DH, (long long)QL * KL, (long long)KL * DH,
                 (long long)QL * DH, 16, NH);

            k_pack_attn<<<(NH * QL * DH) / 256, 256, 0, stream>>>(attn, arow, b);
        }

        // o-projection: (2048 x 1024) @ (1024 x 1024)
        gemm(arow, Wo + (size_t)l * DM * DM, gout,
             QL * BS, DM, DM, DM, DM, DM, 0, 0, 0, 1, 1);
        k_add_ln<<<QL * BS, 256, 0, stream>>>(h, gout, (const float*)nullptr,
                                              ln1g + (size_t)l * DM, ln1b + (size_t)l * DM, hh);

        // FFN
        gemm(hh, Wff1 + (size_t)l * DM * DI, core1,
             QL * BS, DI, DM, DM, DI, DI, 0, 0, 0, 1, 1);
        k_bias_relu_h<<<(QL * BS * DI) / 256, 256, 0, stream>>>(
            core1, ffb1 + (size_t)l * DI, core1h, DI - 1, QL * BS * DI);
        gemm(core1h, Wff2 + (size_t)l * DI * DM, gout,
             QL * BS, DM, DI, DI, DM, DM, 0, 0, 0, 1, 1);
        k_add_ln<<<QL * BS, 256, 0, stream>>>(h, gout, ffb2 + (size_t)l * DM,
                                              ln2g + (size_t)l * DM, ln2b + (size_t)l * DM, hh);
    }

    k_out<<<(QL * BS * DM) / 256, 256, 0, stream>>>(h, (float*)d_out);
}
